// LSTM_56581899158010
// MI455X (gfx1250) — matrix operand order repaired and checked by reading, not compiled, not run on hardware
//
#include <hip/hip_runtime.h>
#include <math.h>

// ---------------------------------------------------------------- types
typedef __attribute__((ext_vector_type(16))) __bf16 v16bf;
typedef __attribute__((ext_vector_type(8)))  __bf16 v8bf;
typedef __attribute__((ext_vector_type(8)))  float  v8f;

#define B_   256
#define T_   512
#define DIN  128
#define DF   256
#define DL   512
#define G4   2048                 // 4*DL
#define MROWS (B_*T_)             // 131072

// LDS staging of the per-block W_h slice: 128 rows x 512 bf16, rows padded to
// 520 elements (1040 B = 260 dwords) so ds_load_b128 lanes stride 4 banks.
#define LDS_ROW   520
#define LDS_ROWS  128
#define LDS_BYTES (LDS_ROWS * LDS_ROW * 2)   // 133,120 B  (< 320 KB WGP limit)

// gfx1250 has a hardware v_tanh_f32 (TRANS32) — use it for all activations.
#if __has_builtin(__builtin_amdgcn_tanhf)
__device__ __forceinline__ float tanh_hw(float x) { return __builtin_amdgcn_tanhf(x); }
#else
__device__ __forceinline__ float tanh_hw(float x) { return tanhf(x); }
#endif
__device__ __forceinline__ float sigmoid_hw(float x) {
  return 0.5f + 0.5f * tanh_hw(0.5f * x);   // single TRANS op instead of libm expf chain
}

// ---------------------------------------------------------------- WMMA K-loop
// A: row-major [M,K] bf16.  BT: B^T row-major [N,K] bf16 (B columns contiguous).
// A-operand (16-bit, 16x32): lanes 0-15 hold row M=lane, K=k0..k0+15 (contiguous);
// lanes 16-31 hold K=k0+16..31.  B-operand: lane n holds col n0+n,
// 16 contiguous K per half-wave.  C/D: col n0+(lane&15), rows m0+r+8*(lane>>4).
__device__ __forceinline__ v8f wmma_kloop(const __bf16* __restrict__ A, int lda,
                                          const __bf16* __restrict__ BT, int ldb,
                                          int m0, int n0, int K, v8f acc)
{
  const int lane = threadIdx.x & 31;
  const int ln   = lane & 15;
  const int hi   = lane >> 4;
  const __bf16* arow = A  + (size_t)(m0 + ln) * lda + hi * 16;
  const __bf16* bcol = BT + (size_t)(n0 + ln) * ldb + hi * 16;
  for (int k0 = 0; k0 < K; k0 += 32) {
    v16bf a, b;
    *(v8bf*)&a        = *(const v8bf*)(arow + k0);
    *((v8bf*)&a + 1)  = *(const v8bf*)(arow + k0 + 8);
    b                 = *(const v16bf*)(bcol + k0);
    acc = __builtin_amdgcn_wmma_f32_16x16x32_bf16(false, a, false, b,
                                                  (short)0, acc, false, false);
  }
  return acc;
}

// ---------------------------------------------------------------- prep kernels
__global__ void zero_init_kernel(__bf16* __restrict__ h, unsigned* __restrict__ sync) {
  int i = blockIdx.x * blockDim.x + threadIdx.x;
  if (i < B_ * DL) h[i] = (__bf16)0.0f;
  if (i < 2) sync[i] = 0u;
}

__global__ void cast_f32_bf16_kernel(const float* __restrict__ in,
                                     __bf16* __restrict__ out, int n) {
  int i = blockIdx.x * blockDim.x + threadIdx.x;
  if (i < n) out[i] = (__bf16)in[i];
}

// out[c][r] = (bf16) in[r][c]   — makes B columns contiguous in K.
__global__ void transpose_cast_kernel(const float* __restrict__ in,
                                      __bf16* __restrict__ out, int R, int C) {
  int i = blockIdx.x * blockDim.x + threadIdx.x;
  if (i < R * C) {
    int r = i / C, c = i % C;
    out[(size_t)c * R + r] = (__bf16)in[i];
  }
}

// ---------------------------------------------------------------- encoder GEMM
// enc = tanh(x @ W_enc + b_enc)  -> bf16 [MROWS, DF]
__global__ void enc_gemm_kernel(const __bf16* __restrict__ Xb,     // [MROWS,DIN]
                                const __bf16* __restrict__ WencT,  // [DF,DIN]
                                const float*  __restrict__ bias,   // [DF]
                                __bf16* __restrict__ enc)          // [MROWS,DF]
{
  const int lane = threadIdx.x & 31;
  const int wave = threadIdx.x >> 5;
  const int ln = lane & 15, hi = lane >> 4;
  const int m0 = blockIdx.x * 128 + wave * 16;
  const int n0 = blockIdx.y * 16;
  v8f acc = {};
  acc = wmma_kloop(Xb, DIN, WencT, DIN, m0, n0, DIN, acc);
  const float bv = bias[n0 + ln];
#pragma unroll
  for (int r = 0; r < 8; ++r) {
    const int row = m0 + r + 8 * hi;
    enc[(size_t)row * DF + n0 + ln] = (__bf16)tanh_hw(acc[r] + bv);
  }
}

// ---------------------------------------------------------------- xz GEMM
// xz[t,b,:] = enc[b*T+t] @ W_x + b_lstm, stored in WMMA C-FRAGMENT layout:
//   frag[t][bt][nt][lane][r]  (bt = batch-tile b/16, nt = gate-col-tile n/16)
// so the recurrent kernel can reload each 16x16 tile as one 16B load per lane.
// Grid: (T, B/128, G4/16); block 256 = 8 waves, each wave = 16 batch rows.
__global__ void xz_gemm_kernel(const __bf16* __restrict__ Enc,   // [MROWS,DF]
                               const __bf16* __restrict__ WxT,   // [G4,DF]
                               const float*  __restrict__ bias,  // [G4]
                               __bf16* __restrict__ xzf)         // fragments
{
  const int t    = blockIdx.x;
  const int lane = threadIdx.x & 31;
  const int wave = threadIdx.x >> 5;
  const int ln = lane & 15, hi = lane >> 4;
  const int b0 = blockIdx.y * 128 + wave * 16;   // batch-tile base
  const int n0 = blockIdx.z * 16;                // gate-column tile base
  v8f acc = {};
  const __bf16* arow = Enc + ((size_t)(b0 + ln) * T_ + t) * DF + hi * 16;
  const __bf16* bcol = WxT + (size_t)(n0 + ln) * DF + hi * 16;
  for (int k0 = 0; k0 < DF; k0 += 32) {
    v16bf a, b;
    *(v8bf*)&a       = *(const v8bf*)(arow + k0);
    *((v8bf*)&a + 1) = *(const v8bf*)(arow + k0 + 8);
    b                = *(const v16bf*)(bcol + k0);
    acc = __builtin_amdgcn_wmma_f32_16x16x32_bf16(false, a, false, b,
                                                  (short)0, acc, false, false);
  }
  const float bv = bias[n0 + ln];
  v8bf st;
#pragma unroll
  for (int r = 0; r < 8; ++r) st[r] = (__bf16)(acc[r] + bv);
  const size_t frag = (((size_t)t * (B_ / 16) + (b0 >> 4)) * (G4 / 16) + (n0 >> 4)) * 256
                      + (size_t)lane * 8;
  *(v8bf*)(xzf + frag) = st;
}

// ---------------------------------------------------------------- grid barrier
__device__ __forceinline__ void grid_barrier(unsigned* cnt, unsigned* gen,
                                             unsigned nblocks) {
  __syncthreads();
  if (threadIdx.x == 0) {
    const unsigned g = __hip_atomic_load(gen, __ATOMIC_RELAXED,
                                         __HIP_MEMORY_SCOPE_AGENT);
    const unsigned arrived = __hip_atomic_fetch_add(cnt, 1u, __ATOMIC_ACQ_REL,
                                                    __HIP_MEMORY_SCOPE_AGENT);
    if (arrived == nblocks - 1) {
      __hip_atomic_store(cnt, 0u, __ATOMIC_RELAXED, __HIP_MEMORY_SCOPE_AGENT);
      __hip_atomic_fetch_add(gen, 1u, __ATOMIC_ACQ_REL, __HIP_MEMORY_SCOPE_AGENT);
    } else {
      while (__hip_atomic_load(gen, __ATOMIC_ACQUIRE,
                               __HIP_MEMORY_SCOPE_AGENT) == g) {
        __builtin_amdgcn_s_sleep(2);
      }
    }
  }
  __syncthreads();
}

// ---------------------------------------------------------------- LSTM scan
// Persistent kernel, 64 blocks x 256 threads (8 waves).  Each wave owns a
// fixed 16(batch) x 16(latent) tile; cell state c stays in 8 VGPRs for all
// 512 steps; the same wave computes all 4 gates of its tile.  The block's
// 128 KB W_h slice (time-invariant) is staged once into LDS; per-step B
// operands come from ds_load, leaving VMEM for h + streaming xz fragments.
__global__ void lstm_scan_kernel(const __bf16* __restrict__ WhT,  // [G4,DL]
                                 const __bf16* __restrict__ xzf,  // fragments
                                 __bf16* __restrict__ h,          // [B][DL]
                                 unsigned* __restrict__ sync)     // cnt, gen
{
  extern __shared__ __bf16 lwh[];                   // [128][LDS_ROW]

  const int lane = threadIdx.x & 31;
  const int wave = threadIdx.x >> 5;
  const int ln = lane & 15, hi = lane >> 4;
  const int mblk = blockIdx.x >> 4;                 // 0..3  (batch groups of 64)
  const int jblk = blockIdx.x & 15;                 // 0..15 (latent groups of 32)
  const int m0 = mblk * 64 + (wave & 3) * 16;       // batch-row base
  const int j0 = jblk * 32 + (wave >> 2) * 16;      // latent-col base
  const int jloc = j0 & 31;                         // 0 or 16 within block slice

  // ---- one-time stage of this block's W_h slice into LDS --------------
  // LDS row (g*32 + rr) <- WhT row (g*512 + jblk*32 + rr), 512 bf16 each.
  for (int ch = threadIdx.x; ch < LDS_ROWS * 64; ch += 256) {  // 16B segments
    const int row = ch >> 6;                // 0..127
    const int seg = ch & 63;                // 64 x 16B per row
    const int g = row >> 5, rr = row & 31;
    const v8bf v = *(const v8bf*)(WhT + (size_t)(g * DL + jblk * 32 + rr) * DL + seg * 8);
    *(v8bf*)(lwh + (size_t)row * LDS_ROW + seg * 8) = v;
  }
  __syncthreads();

  // LDS B-operand base pointers for the 4 gates of this wave's columns
  const __bf16* lbi = lwh + (size_t)(0 * 32 + jloc + ln) * LDS_ROW + hi * 16;
  const __bf16* lbf = lwh + (size_t)(1 * 32 + jloc + ln) * LDS_ROW + hi * 16;
  const __bf16* lbg = lwh + (size_t)(2 * 32 + jloc + ln) * LDS_ROW + hi * 16;
  const __bf16* lbo = lwh + (size_t)(3 * 32 + jloc + ln) * LDS_ROW + hi * 16;
  const __bf16* arow = h + (size_t)(m0 + ln) * DL + hi * 16;

  // fragment tile offsets for the 4 gates of this wave's latent columns
  const size_t ntI = (size_t)((0 * DL + j0) >> 4) * 256;
  const size_t ntF = (size_t)((1 * DL + j0) >> 4) * 256;
  const size_t ntG = (size_t)((2 * DL + j0) >> 4) * 256;
  const size_t ntO = (size_t)((3 * DL + j0) >> 4) * 256;
  const size_t laneOff = (size_t)lane * 8;

  v8f c = {};                                       // cell state, VGPR-resident
  for (int t = 0; t < T_; ++t) {
    // prefetch next step's xz fragments while this step computes
    if (t + 1 < T_) {
      const __bf16* nf = xzf + ((size_t)(t + 1) * (B_ / 16) + (m0 >> 4)) * (G4 / 16) * 256
                         + laneOff;
      __builtin_prefetch(nf + ntI, 0, 1);
      __builtin_prefetch(nf + ntF, 0, 1);
      __builtin_prefetch(nf + ntG, 0, 1);
      __builtin_prefetch(nf + ntO, 0, 1);
    }
    v8f ai = {}, af = {}, ag = {}, ao = {};
#pragma unroll 4
    for (int k0 = 0; k0 < DL; k0 += 32) {
      v16bf a, wbi, wbf, wbg, wbo;
      *(v8bf*)&a       = *(const v8bf*)(arow + k0);    // h tile: global (L2)
      *((v8bf*)&a + 1) = *(const v8bf*)(arow + k0 + 8);
      wbi = *(const v16bf*)(lbi + k0);                 // W_h tiles: LDS
      wbf = *(const v16bf*)(lbf + k0);
      wbg = *(const v16bf*)(lbg + k0);
      wbo = *(const v16bf*)(lbo + k0);
      ai = __builtin_amdgcn_wmma_f32_16x16x32_bf16(false, a, false, wbi, (short)0, ai, false, false);
      af = __builtin_amdgcn_wmma_f32_16x16x32_bf16(false, a, false, wbf, (short)0, af, false, false);
      ag = __builtin_amdgcn_wmma_f32_16x16x32_bf16(false, a, false, wbg, (short)0, ag, false, false);
      ao = __builtin_amdgcn_wmma_f32_16x16x32_bf16(false, a, false, wbo, (short)0, ao, false, false);
    }
    // xz fragments for this (t, batch-tile): one 16B load per gate
    const __bf16* fbase = xzf + ((size_t)t * (B_ / 16) + (m0 >> 4)) * (G4 / 16) * 256
                          + laneOff;
    const v8bf xi = *(const v8bf*)(fbase + ntI);
    const v8bf xf = *(const v8bf*)(fbase + ntF);
    const v8bf xg = *(const v8bf*)(fbase + ntG);
    const v8bf xo = *(const v8bf*)(fbase + ntO);
#pragma unroll
    for (int r = 0; r < 8; ++r) {
      const int row = m0 + r + 8 * hi;              // batch index
      const float iv = sigmoid_hw(ai[r] + (float)xi[r]);
      const float fv = sigmoid_hw(af[r] + (float)xf[r]);
      const float gv = tanh_hw   (ag[r] + (float)xg[r]);
      const float ov = sigmoid_hw(ao[r] + (float)xo[r]);
      const float cv = fv * c[r] + iv * gv;
      c[r] = cv;
      h[(size_t)row * DL + j0 + ln] = (__bf16)(ov * tanh_hw(cv));
    }
    grid_barrier(sync, sync + 1, gridDim.x);
  }
}

// ---------------------------------------------------------------- decoder
__global__ void decoder_kernel(const __bf16* __restrict__ h,  // [B][DL]
                               const float* __restrict__ Wd,  // [DL]
                               const float* __restrict__ bd,  // [1]
                               float* __restrict__ out)       // [B]
{
  __shared__ float red[256];
  const int b = blockIdx.x;
  float s = 0.0f;
  for (int d = threadIdx.x; d < DL; d += 256)
    s += (float)h[(size_t)b * DL + d] * Wd[d];
  red[threadIdx.x] = s;
  __syncthreads();
  for (int off = 128; off > 0; off >>= 1) {
    if (threadIdx.x < off) red[threadIdx.x] += red[threadIdx.x + off];
    __syncthreads();
  }
  if (threadIdx.x == 0) out[b] = red[0] + bd[0];
}

// ---------------------------------------------------------------- launch
extern "C" void kernel_launch(void* const* d_in, const int* in_sizes, int n_in,
                              void* d_out, int out_size, void* d_ws, size_t ws_size,
                              hipStream_t stream) {
  (void)in_sizes; (void)n_in; (void)out_size; (void)ws_size;
  const float* x      = (const float*)d_in[0];  // [B,T,DIN]
  const float* W_enc  = (const float*)d_in[1];  // [DIN,DF]
  const float* b_enc  = (const float*)d_in[2];  // [DF]
  const float* W_x    = (const float*)d_in[3];  // [DF,G4]
  const float* W_h    = (const float*)d_in[4];  // [DL,G4]
  const float* b_lstm = (const float*)d_in[5];  // [G4]
  const float* W_dec  = (const float*)d_in[6];  // [DL]
  const float* b_dec  = (const float*)d_in[7];  // [1]
  float* out = (float*)d_out;

  char* ws = (char*)d_ws;
  size_t off = 0;
  __bf16* xb    = (__bf16*)(ws + off); off += (size_t)MROWS * DIN * 2;   //  33.5 MB
  __bf16* enc   = (__bf16*)(ws + off); off += (size_t)MROWS * DF  * 2;   //  67.1 MB
  __bf16* WencT = (__bf16*)(ws + off); off += (size_t)DF * DIN * 2;      //  64 KB
  __bf16* WxT   = (__bf16*)(ws + off); off += (size_t)G4 * DF  * 2;      //   1 MB
  __bf16* WhT   = (__bf16*)(ws + off); off += (size_t)G4 * DL  * 2;      //   2 MB
  __bf16* xzf   = (__bf16*)(ws + off); off += (size_t)T_ * B_ * G4 * 2;  // 537 MB
  __bf16* hbuf  = (__bf16*)(ws + off); off += (size_t)B_ * DL * 2;       // 256 KB
  unsigned* sync = (unsigned*)(ws + off);

  // 1) zero h + barrier state (ws is poisoned by harness)
  zero_init_kernel<<<(B_ * DL + 255) / 256, 256, 0, stream>>>(hbuf, sync);
  // 2) cast x -> bf16
  cast_f32_bf16_kernel<<<(MROWS * DIN + 255) / 256, 256, 0, stream>>>(x, xb, MROWS * DIN);
  // 3) transpose+cast weights so WMMA B-operands are contiguous in K
  transpose_cast_kernel<<<(DIN * DF + 255) / 256, 256, 0, stream>>>(W_enc, WencT, DIN, DF);
  transpose_cast_kernel<<<(DF * G4 + 255) / 256, 256, 0, stream>>>(W_x, WxT, DF, G4);
  transpose_cast_kernel<<<(DL * G4 + 255) / 256, 256, 0, stream>>>(W_h, WhT, DL, G4);
  // 4) encoder GEMM (tanh fused)
  enc_gemm_kernel<<<dim3(MROWS / 128, DF / 16), 256, 0, stream>>>(xb, WencT, b_enc, enc);
  // 5) xz GEMM, output in WMMA C-fragment layout [t][bt][nt][lane][r]
  xz_gemm_kernel<<<dim3(T_, B_ / 128, G4 / 16), 256, 0, stream>>>(enc, WxT, b_lstm, xzf);
  // 6) persistent LSTM recurrence: 64 blocks, W_h slice LDS-resident,
  //    grid barrier per timestep
  lstm_scan_kernel<<<64, 256, LDS_BYTES, stream>>>(WhT, xzf, hbuf, sync);
  // 7) decoder on h_last
  decoder_kernel<<<B_, 256, 0, stream>>>(hbuf, W_dec, b_dec, out);
}
